// CrystalGraphDiffusionModel_81484119540315
// MI455X (gfx1250) — matrix operand order repaired and checked by reading, not compiled, not run on hardware
//
#include <hip/hip_runtime.h>
#include <math.h>

#define NN 10000
#define EE 160000
#define GG 16

typedef __attribute__((ext_vector_type(16))) _Float16 v16h;
typedef __attribute__((ext_vector_type(8)))  _Float16 v8h;
typedef __attribute__((ext_vector_type(8)))  float    v8f;

// ---------- fragment helpers (f16 operands, f32 accumulate) ----------
// A 16x32 fragment (ISA 7.12.2): lanes 0-15 -> K = k0+{0..7,16..23},
// lanes 16-31 -> K = k0+{8..15,24..31}; row M = lane&15. rowp is an f16 row.
__device__ __forceinline__ v16h frag_a16(const _Float16* rowp, int kb) {
  v8h lo = *(const v8h*)(rowp + kb);
  v8h hi = *(const v8h*)(rowp + kb + 16);
  return __builtin_shufflevector(lo, hi, 0,1,2,3,4,5,6,7,8,9,10,11,12,13,14,15);
}
// B 32x16 fragment from pre-swizzled weights Wz[k0/32][col][32]:
// lanes 0-15 take K=k0+{0..7,16..23}, lanes 16-31 take K=k0+{8..15,24..31} (matches A).
__device__ __forceinline__ v16h frag_bz(const _Float16* Wz, int Nout, int col, int k0, int sel) {
  const _Float16* p = Wz + (((size_t)(k0 >> 5) * Nout + col) << 5) + (sel << 3);
  v8h lo = *(const v8h*)p;
  v8h hi = *(const v8h*)(p + 16);
  return __builtin_shufflevector(lo, hi, 0,1,2,3,4,5,6,7,8,9,10,11,12,13,14,15);
}
__device__ __forceinline__ v8f wmma_step(v16h a, v16h b, v8f c) {
  return __builtin_amdgcn_wmma_f32_16x16x32_f16(false, a, false, b, (short)0, c, false, false);
}
__device__ __forceinline__ float silu_f(float x) {
  return x * __builtin_amdgcn_rcpf(1.f + __expf(-x));   // v_rcp_f32, no IEEE div chain
}
__device__ __forceinline__ unsigned fenc(float x) {
  unsigned u = __float_as_uint(x);
  return (u & 0x80000000u) ? ~u : (u | 0x80000000u);
}
__device__ __forceinline__ float fdec(unsigned u) {
  return (u & 0x80000000u) ? __uint_as_float(u & 0x7fffffffu) : __uint_as_float(~u);
}

// ---------- one-time converts / weight swizzle ----------
__global__ void cvt_kernel(const float* __restrict__ X, _Float16* __restrict__ Y, int n) {
  int i = blockIdx.x * blockDim.x + threadIdx.x;
  if (i < n) Y[i] = (_Float16)X[i];
}
__global__ void wswz_kernel(const float* __restrict__ W, _Float16* __restrict__ Wz,
                            int K, int Nout) {
  int i = blockIdx.x * blockDim.x + threadIdx.x;
  if (i >= K * Nout) return;
  int k = i / Nout, n = i - k * Nout;
  Wz[((size_t)(k >> 5) * Nout + n) * 32 + (k & 31)] = (_Float16)W[i];
}

// ---------- generic WMMA GEMM, 16x32 output per wave ----------
template <int ACT>
__global__ void gemm16_kernel(const _Float16* __restrict__ A, const _Float16* __restrict__ Wz,
                              const float* __restrict__ bias,
                              float* __restrict__ Cf, _Float16* __restrict__ Ch,
                              int M, int K, int Nout) {
  int wid = threadIdx.x >> 5, lane = threadIdx.x & 31;
  int tilesN = Nout >> 5;
  int total = (M >> 4) * tilesN;
  int tile = blockIdx.x * 8 + wid;
  if (tile >= total) return;                 // wave-uniform
  int tm = tile / tilesN, tn = tile - tm * tilesN;
  int sel = lane >> 4, m = lane & 15;
  int col0 = tn * 32 + m, col1 = col0 + 16;
  const _Float16* arow = A + (size_t)(tm * 16 + m) * K;
  v8f acc0 = {}, acc1 = {};
  for (int k0 = 0; k0 < K; k0 += 32) {
    v16h a = frag_a16(arow, k0 + sel * 8);
    acc0 = wmma_step(a, frag_bz(Wz, Nout, col0, k0, sel), acc0);
    acc1 = wmma_step(a, frag_bz(Wz, Nout, col1, k0, sel), acc1);
  }
  float b0 = bias[col0], b1 = bias[col1];
#pragma unroll
  for (int r = 0; r < 8; r++) {
    size_t row = (size_t)(tm * 16 + r + sel * 8) * Nout;
    float v0 = acc0[r] + b0, v1 = acc1[r] + b1;
    if (ACT == 1) { v0 = silu_f(v0); v1 = silu_f(v1); }
    if (Cf) { Cf[row + col0] = v0; Cf[row + col1] = v1; }
    if (Ch) { Ch[row + col0] = (_Float16)v0; Ch[row + col1] = (_Float16)v1; }
  }
}

// ---------- concat GEMM: silu(concat(A0,A1)[M,512] @ W) -> f16 ----------
__global__ void gemm_cat_silu_kernel(const _Float16* __restrict__ A0,
                                     const _Float16* __restrict__ A1,
                                     const _Float16* __restrict__ Wz,
                                     const float* __restrict__ bias,
                                     _Float16* __restrict__ Ch, int M) {
  int wid = threadIdx.x >> 5, lane = threadIdx.x & 31;
  int total = (M >> 4) * 8;
  int tile = blockIdx.x * 8 + wid;
  if (tile >= total) return;
  int tm = tile / 8, tn = tile - tm * 8;
  int sel = lane >> 4, m = lane & 15;
  int col0 = tn * 32 + m, col1 = col0 + 16;
  int row = tm * 16 + m;
  v8f acc0 = {}, acc1 = {};
  for (int k0 = 0; k0 < 512; k0 += 32) {
    const _Float16* arow = (k0 < 256) ? (A0 + (size_t)row * 256) : (A1 + (size_t)row * 256);
    v16h a = frag_a16(arow, (k0 & 255) + sel * 8);
    acc0 = wmma_step(a, frag_bz(Wz, 256, col0, k0, sel), acc0);
    acc1 = wmma_step(a, frag_bz(Wz, 256, col1, k0, sel), acc1);
  }
  float b0 = bias[col0], b1 = bias[col1];
#pragma unroll
  for (int r = 0; r < 8; r++) {
    size_t ro = (size_t)(tm * 16 + r + sel * 8) * 256;
    Ch[ro + col0] = (_Float16)silu_f(acc0[r] + b0);
    Ch[ro + col1] = (_Float16)silu_f(acc1[r] + b1);
  }
}

// ---------- fused edge pipeline (16 edges / block, intermediates in f16 LDS) ----------
__global__ void edge_fused_kernel(const _Float16* __restrict__ ea16,
                                  const int* __restrict__ eidx,
                                  const _Float16* __restrict__ hn16,
                                  const _Float16* __restrict__ ep1z, const float* __restrict__ ep1b,
                                  const _Float16* __restrict__ ep2z, const float* __restrict__ ep2b,
                                  const _Float16* __restrict__ m1z, const float* __restrict__ m1b,
                                  const _Float16* __restrict__ m2z, const float* __restrict__ m2b,
                                  float* __restrict__ s_out) {
  __shared__ _Float16 sA[16 * 256];
  __shared__ _Float16 sHe[16 * 256];
  __shared__ int sidx[32];  // [0:16) src, [16:32) dst
  int tid = threadIdx.x, wid = tid >> 5, lane = tid & 31;
  int e0 = blockIdx.x * 16;
  if (tid < 16) sidx[tid] = eidx[e0 + tid];
  else if (tid < 32) sidx[tid] = eidx[EE + e0 + tid - 16];
  __syncthreads();
  int sel = lane >> 4, m = lane & 15;
  int col0 = wid * 32 + m, col1 = col0 + 16;   // each wave owns one 32-wide strip
  v8f acc0, acc1;
  // Stage 1: sA = silu(edge_attr_tile @ ep1 + b)   K=64
  acc0 = {}; acc1 = {};
  {
    const _Float16* arow = ea16 + (size_t)(e0 + m) * 64;
    for (int k0 = 0; k0 < 64; k0 += 32) {
      v16h a = frag_a16(arow, k0 + sel * 8);
      acc0 = wmma_step(a, frag_bz(ep1z, 256, col0, k0, sel), acc0);
      acc1 = wmma_step(a, frag_bz(ep1z, 256, col1, k0, sel), acc1);
    }
  }
  {
    float b0 = ep1b[col0], b1 = ep1b[col1];
#pragma unroll
    for (int r = 0; r < 8; r++) {
      int ro = (r + sel * 8) * 256;
      sA[ro + col0] = (_Float16)silu_f(acc0[r] + b0);
      sA[ro + col1] = (_Float16)silu_f(acc1[r] + b1);
    }
  }
  __syncthreads();
  // Stage 2: sHe = sA @ ep2 + b   K=256
  acc0 = {}; acc1 = {};
  for (int k0 = 0; k0 < 256; k0 += 32) {
    v16h a = frag_a16(sA + m * 256, k0 + sel * 8);
    acc0 = wmma_step(a, frag_bz(ep2z, 256, col0, k0, sel), acc0);
    acc1 = wmma_step(a, frag_bz(ep2z, 256, col1, k0, sel), acc1);
  }
  {
    float b0 = ep2b[col0], b1 = ep2b[col1];
#pragma unroll
    for (int r = 0; r < 8; r++) {
      int ro = (r + sel * 8) * 256;
      sHe[ro + col0] = (_Float16)(acc0[r] + b0);
      sHe[ro + col1] = (_Float16)(acc1[r] + b1);
    }
  }
  __syncthreads();
  // Stage 3: sA = silu([hn[dst], hn[src], sHe] @ msg1 + b)   K=768
  acc0 = {}; acc1 = {};
  for (int k0 = 0; k0 < 768; k0 += 32) {
    const _Float16* arow; int kk;
    if (k0 < 256)      { arow = hn16 + (size_t)sidx[16 + m] * 256; kk = k0; }
    else if (k0 < 512) { arow = hn16 + (size_t)sidx[m] * 256;      kk = k0 - 256; }
    else               { arow = sHe + m * 256;                     kk = k0 - 512; }
    v16h a = frag_a16(arow, kk + sel * 8);
    acc0 = wmma_step(a, frag_bz(m1z, 256, col0, k0, sel), acc0);
    acc1 = wmma_step(a, frag_bz(m1z, 256, col1, k0, sel), acc1);
  }
  __syncthreads();   // everyone done reading sHe/old sA
  {
    float b0 = m1b[col0], b1 = m1b[col1];
#pragma unroll
    for (int r = 0; r < 8; r++) {
      int ro = (r + sel * 8) * 256;
      sA[ro + col0] = (_Float16)silu_f(acc0[r] + b0);
      sA[ro + col1] = (_Float16)silu_f(acc1[r] + b1);
    }
  }
  __syncthreads();
  // Stage 4: m = sA @ msg2 + b; scatter-add into s_out[dst]
  acc0 = {}; acc1 = {};
  for (int k0 = 0; k0 < 256; k0 += 32) {
    v16h a = frag_a16(sA + m * 256, k0 + sel * 8);
    acc0 = wmma_step(a, frag_bz(m2z, 256, col0, k0, sel), acc0);
    acc1 = wmma_step(a, frag_bz(m2z, 256, col1, k0, sel), acc1);
  }
  {
    float b0 = m2b[col0], b1 = m2b[col1];
#pragma unroll
    for (int r = 0; r < 8; r++) {
      int row = r + sel * 8;
      size_t dbase = (size_t)sidx[16 + row] * 256;
      atomicAdd(&s_out[dbase + col0], acc0[r] + b0);
      atomicAdd(&s_out[dbase + col1], acc1[r] + b1);
    }
  }
}

// ---------- elementwise / reduction kernels ----------
__global__ void time_embed_kernel(const float* __restrict__ t, float* __restrict__ traw) {
  int i = blockIdx.x * blockDim.x + threadIdx.x;
  if (i >= GG * 128) return;
  int g = i >> 7, j = i & 127;
  float f = __expf(-(float)j * (logf(10000.f) / 127.f));
  float a = t[g] * f;
  traw[g * 256 + j] = sinf(a);
  traw[g * 256 + 128 + j] = cosf(a);
}

__global__ void add_graph_emb(float* __restrict__ h, _Float16* __restrict__ h16,
                              const float* __restrict__ temb, const float* __restrict__ cemb,
                              const int* __restrict__ batch) {
  int i = blockIdx.x * blockDim.x + threadIdx.x;
  if (i >= NN * 256) return;
  int row = i >> 8, col = i & 255;
  int g = batch[row];
  float v = h[i] + temb[g * 256 + col] + cemb[g * 256 + col];
  h[i] = v;
  h16[i] = (_Float16)v;
}

__global__ void count_kernel(const int* __restrict__ eidx, float* __restrict__ cnt) {
  int i = blockIdx.x * blockDim.x + threadIdx.x;
  if (i >= EE) return;
  atomicAdd(&cnt[eidx[EE + i]], 1.f);
}

__global__ void mean_kernel(float* __restrict__ aggr, const float* __restrict__ cnt,
                            _Float16* __restrict__ aggr16) {
  int i = blockIdx.x * blockDim.x + threadIdx.x;
  if (i >= NN * 256) return;
  float v = aggr[i] * __builtin_amdgcn_rcpf(fmaxf(cnt[i >> 8], 1.f));
  aggr[i] = v;
  aggr16[i] = (_Float16)v;
}

__global__ void ln_kernel(const float* __restrict__ X, const float* __restrict__ R,
                          const float* __restrict__ g, const float* __restrict__ b,
                          float* __restrict__ Yf, _Float16* __restrict__ Yh) {
  __shared__ float sv[256], sq[256];
  int t = threadIdx.x; int row = blockIdx.x;
  float x = X[(size_t)row * 256 + t];
  if (R) x += R[(size_t)row * 256 + t];
  sv[t] = x; sq[t] = x * x;
  __syncthreads();
  for (int s = 128; s > 0; s >>= 1) {
    if (t < s) { sv[t] += sv[t + s]; sq[t] += sq[t + s]; }
    __syncthreads();
  }
  float mean = sv[0] * (1.f / 256.f);
  float var = sq[0] * (1.f / 256.f) - mean * mean;
  float inv = __builtin_amdgcn_rsqf(var + 1e-5f);
  float y = (x - mean) * inv * g[t] + b[t];
  Yf[(size_t)row * 256 + t] = y;
  if (Yh) Yh[(size_t)row * 256 + t] = (_Float16)y;
}

__global__ void attn_score_kernel(const float* __restrict__ q, const float* __restrict__ k,
                                  const float* __restrict__ edge_attr,
                                  const float* __restrict__ ejw, const float* __restrict__ ejb,
                                  const int* __restrict__ eidx,
                                  float* __restrict__ scores, unsigned* __restrict__ smax) {
  int i = blockIdx.x * blockDim.x + threadIdx.x;
  if (i >= EE * 8) return;
  int e = i >> 3, hh = i & 7;
  int s = eidx[e], d = eidx[EE + e];
  const float* qp = q + (size_t)d * 256 + hh * 32;
  const float* kp = k + (size_t)s * 256 + hh * 32;
  float acc = 0.f;
#pragma unroll
  for (int j = 0; j < 32; j++) acc += qp[j] * kp[j];
  float ew = ejb[hh];
  const float* ea = edge_attr + (size_t)e * 64;
#pragma unroll 8
  for (int j = 0; j < 64; j++) ew += ea[j] * ejw[j * 8 + hh];
  float sc = acc * 0.17677669529663687f + ew;  // 32^-0.5
  scores[i] = sc;
  atomicMax(&smax[d * 8 + hh], fenc(sc));
}

__global__ void attn_exp_kernel(const float* __restrict__ scores, const unsigned* __restrict__ smax,
                                const int* __restrict__ eidx,
                                float* __restrict__ exb, float* __restrict__ denom) {
  int i = blockIdx.x * blockDim.x + threadIdx.x;
  if (i >= EE * 8) return;
  int e = i >> 3, hh = i & 7;
  int d = eidx[EE + e];
  float ex = __expf(scores[i] - fdec(smax[d * 8 + hh]));
  exb[i] = ex;
  atomicAdd(&denom[d * 8 + hh], ex);
}

__global__ void attn_out_kernel(const float* __restrict__ v, const float* __restrict__ exb,
                                const float* __restrict__ denom, const int* __restrict__ eidx,
                                float* __restrict__ attnout) {
  int i = blockIdx.x * blockDim.x + threadIdx.x;
  if (i >= EE * 256) return;
  int e = i >> 8, r = i & 255, hh = r >> 5;
  int s = eidx[e], d = eidx[EE + e];
  float alpha = exb[e * 8 + hh] * __builtin_amdgcn_rcpf(fmaxf(denom[d * 8 + hh], 1e-20f));
  atomicAdd(&attnout[(size_t)d * 256 + r], v[(size_t)s * 256 + r] * alpha);
}

// ---------- host launcher ----------
extern "C" void kernel_launch(void* const* d_in, const int* in_sizes, int n_in,
                              void* d_out, int out_size, void* d_ws, size_t ws_size,
                              hipStream_t stream) {
  (void)in_sizes; (void)n_in; (void)out_size; (void)ws_size;
  auto F = [&](int i) { return (const float*)d_in[i]; };
  const float* x = F(0);
  const int* eidx = (const int*)d_in[1];
  const float* edge_attr = F(2);
  const float* t = F(3);
  const float* cond = F(4);
  const int* batch = (const int*)d_in[5];
  int p = 6;
  const float *enc_w = F(p++), *enc_b = F(p++);
  const float *tm1_w = F(p++), *tm1_b = F(p++);
  const float *tm2_w = F(p++), *tm2_b = F(p++);
  const float *cond_w = F(p++), *cond_b = F(p++);
  const float *out1_w = F(p++), *out1_b = F(p++);
  const float *out2_w = F(p++), *out2_b = F(p++);
  struct Lyr {
    const float *np1w,*np1b,*np2w,*np2b,*ep1w,*ep1b,*ep2w,*ep2b;
    const float *m1w,*m1b,*m2w,*m2b,*u1w,*u1b,*u2w,*u2b;
    const float *lng,*lnb,*qw,*qb2,*kw,*kb2,*vw,*vb2,*ejw,*ejb,*ow,*ob,*alng,*alnb;
  } Ls[3];
  for (int l = 0; l < 3; l++) {
    Lyr& L = Ls[l];
    L.np1w=F(p++); L.np1b=F(p++); L.np2w=F(p++); L.np2b=F(p++);
    L.ep1w=F(p++); L.ep1b=F(p++); L.ep2w=F(p++); L.ep2b=F(p++);
    L.m1w=F(p++);  L.m1b=F(p++);  L.m2w=F(p++);  L.m2b=F(p++);
    L.u1w=F(p++);  L.u1b=F(p++);  L.u2w=F(p++);  L.u2b=F(p++);
    L.lng=F(p++);  L.lnb=F(p++);
    L.qw=F(p++); L.qb2=F(p++); L.kw=F(p++); L.kb2=F(p++); L.vw=F(p++); L.vb2=F(p++);
    L.ejw=F(p++); L.ejb=F(p++); L.ow=F(p++); L.ob=F(p++); L.alng=F(p++); L.alnb=F(p++);
  }

  // workspace bump allocator
  char* base = (char*)d_ws;
  size_t off = 0;
  auto alloc = [&](size_t bytes) { void* r = base + off; off += (bytes + 255) & ~(size_t)255; return r; };
  // f32 buffers
  float* h       = (float*)alloc((size_t)NN * 256 * 4);
  float* aggr    = (float*)alloc((size_t)NN * 256 * 4);
  float* qbuf    = (float*)alloc((size_t)NN * 256 * 4);
  float* kbuf    = (float*)alloc((size_t)NN * 256 * 4);
  float* vbuf    = (float*)alloc((size_t)NN * 256 * 4);
  float* attnout = (float*)alloc((size_t)NN * 256 * 4);
  float* cnt     = (float*)alloc((size_t)NN * 4);
  float* scores  = (float*)alloc((size_t)EE * 8 * 4);
  float* exb     = (float*)alloc((size_t)EE * 8 * 4);
  unsigned* smax = (unsigned*)alloc((size_t)NN * 8 * 4);
  float* denom   = (float*)alloc((size_t)NN * 8 * 4);
  float* traw    = (float*)alloc((size_t)GG * 256 * 4);
  float* temb    = (float*)alloc((size_t)GG * 256 * 4);
  float* cemb    = (float*)alloc((size_t)GG * 256 * 4);
  // f16 activation buffers
  _Float16* h16    = (_Float16*)alloc((size_t)NN * 256 * 2);
  _Float16* ntmp16 = (_Float16*)alloc((size_t)NN * 256 * 2);
  _Float16* hn16   = (_Float16*)alloc((size_t)NN * 256 * 2);
  _Float16* aggr16 = (_Float16*)alloc((size_t)NN * 256 * 2);
  _Float16* x16    = (_Float16*)alloc((size_t)NN * 128 * 2);
  _Float16* ea16   = (_Float16*)alloc((size_t)EE * 64 * 2);
  _Float16* cond16 = (_Float16*)alloc((size_t)GG * 32 * 2);
  _Float16* traw16 = (_Float16*)alloc((size_t)GG * 256 * 2);
  _Float16* tmid16 = (_Float16*)alloc((size_t)GG * 256 * 2);
  // f16 swizzled weights
  auto allocz = [&](int K, int Nout) { return (_Float16*)alloc((size_t)K * Nout * 2); };
  _Float16* enc_z = allocz(128, 256);
  _Float16* tm1_z = allocz(256, 256);
  _Float16* tm2_z = allocz(256, 256);
  _Float16* cond_z = allocz(32, 256);
  _Float16* out1_z = allocz(256, 256);
  _Float16* out2_z = allocz(256, 128);
  struct LyrZ { _Float16 *np1,*np2,*ep1,*ep2,*m1,*m2,*u1,*u2,*q,*k,*v,*o; } Lz[3];
  for (int l = 0; l < 3; l++) {
    Lz[l].np1 = allocz(256, 256); Lz[l].np2 = allocz(256, 256);
    Lz[l].ep1 = allocz(64, 256);  Lz[l].ep2 = allocz(256, 256);
    Lz[l].m1  = allocz(768, 256); Lz[l].m2  = allocz(256, 256);
    Lz[l].u1  = allocz(512, 256); Lz[l].u2  = allocz(256, 256);
    Lz[l].q   = allocz(256, 256); Lz[l].k   = allocz(256, 256);
    Lz[l].v   = allocz(256, 256); Lz[l].o   = allocz(256, 256);
  }

  auto swz = [&](const float* W, _Float16* Wz, int K, int Nout) {
    wswz_kernel<<<((K * Nout) + 255) / 256, 256, 0, stream>>>(W, Wz, K, Nout);
  };
  auto cvt = [&](const float* X, _Float16* Y, int n) {
    cvt_kernel<<<(n + 255) / 256, 256, 0, stream>>>(X, Y, n);
  };
  auto gblk = [](int M, int Nout) { return ((M >> 4) * (Nout >> 5) + 7) / 8; };

  // --- one-time converts / swizzles ---
  cvt(x, x16, NN * 128);
  cvt(edge_attr, ea16, EE * 64);
  cvt(cond, cond16, GG * 32);
  swz(enc_w, enc_z, 128, 256);
  swz(tm1_w, tm1_z, 256, 256);  swz(tm2_w, tm2_z, 256, 256);
  swz(cond_w, cond_z, 32, 256);
  swz(out1_w, out1_z, 256, 256); swz(out2_w, out2_z, 256, 128);
  for (int l = 0; l < 3; l++) {
    const Lyr& L = Ls[l];  LyrZ& Z = Lz[l];
    swz(L.np1w, Z.np1, 256, 256); swz(L.np2w, Z.np2, 256, 256);
    swz(L.ep1w, Z.ep1, 64, 256);  swz(L.ep2w, Z.ep2, 256, 256);
    swz(L.m1w,  Z.m1, 768, 256);  swz(L.m2w,  Z.m2, 256, 256);
    swz(L.u1w,  Z.u1, 512, 256);  swz(L.u2w,  Z.u2, 256, 256);
    swz(L.qw, Z.q, 256, 256); swz(L.kw, Z.k, 256, 256);
    swz(L.vw, Z.v, 256, 256); swz(L.ow, Z.o, 256, 256);
  }

  // --- embeddings + encoder ---
  time_embed_kernel<<<(GG * 128 + 255) / 256, 256, 0, stream>>>(t, traw);
  cvt(traw, traw16, GG * 256);
  gemm16_kernel<1><<<gblk(GG, 256), 256, 0, stream>>>(traw16, tm1_z, tm1_b, nullptr, tmid16, GG, 256, 256);
  gemm16_kernel<0><<<gblk(GG, 256), 256, 0, stream>>>(tmid16, tm2_z, tm2_b, temb, nullptr, GG, 256, 256);
  gemm16_kernel<0><<<gblk(GG, 256), 256, 0, stream>>>(cond16, cond_z, cond_b, cemb, nullptr, GG, 32, 256);
  gemm16_kernel<0><<<gblk(NN, 256), 256, 0, stream>>>(x16, enc_z, enc_b, h, nullptr, NN, 128, 256);
  add_graph_emb<<<(NN * 256 + 255) / 256, 256, 0, stream>>>(h, h16, temb, cemb, batch);

  // in-degree counts (constant across layers)
  hipMemsetAsync(cnt, 0, (size_t)NN * 4, stream);
  count_kernel<<<(EE + 255) / 256, 256, 0, stream>>>(eidx, cnt);

  for (int l = 0; l < 3; l++) {
    const Lyr& L = Ls[l];  const LyrZ& Z = Lz[l];
    // --- EquivariantGraphConv ---
    gemm16_kernel<1><<<gblk(NN, 256), 256, 0, stream>>>(h16, Z.np1, L.np1b, nullptr, ntmp16, NN, 256, 256);
    gemm16_kernel<0><<<gblk(NN, 256), 256, 0, stream>>>(ntmp16, Z.np2, L.np2b, nullptr, hn16, NN, 256, 256);
    hipMemsetAsync(aggr, 0, (size_t)NN * 256 * 4, stream);
    edge_fused_kernel<<<EE / 16, 256, 0, stream>>>(ea16, eidx, hn16,
        Z.ep1, L.ep1b, Z.ep2, L.ep2b, Z.m1, L.m1b, Z.m2, L.m2b, aggr);
    mean_kernel<<<(NN * 256 + 255) / 256, 256, 0, stream>>>(aggr, cnt, aggr16);
    gemm_cat_silu_kernel<<<gblk(NN, 256), 256, 0, stream>>>(h16, aggr16, Z.u1, L.u1b, ntmp16, NN);
    gemm16_kernel<0><<<gblk(NN, 256), 256, 0, stream>>>(ntmp16, Z.u2, L.u2b, qbuf, nullptr, NN, 256, 256); // qbuf := upd
    ln_kernel<<<NN, 256, 0, stream>>>(h, qbuf, L.lng, L.lnb, h, h16);
    // --- GraphAttention ---
    gemm16_kernel<0><<<gblk(NN, 256), 256, 0, stream>>>(h16, Z.q, L.qb2, qbuf, nullptr, NN, 256, 256);
    gemm16_kernel<0><<<gblk(NN, 256), 256, 0, stream>>>(h16, Z.k, L.kb2, kbuf, nullptr, NN, 256, 256);
    gemm16_kernel<0><<<gblk(NN, 256), 256, 0, stream>>>(h16, Z.v, L.vb2, vbuf, nullptr, NN, 256, 256);
    hipMemsetAsync(smax, 0, (size_t)NN * 8 * 4, stream);
    hipMemsetAsync(denom, 0, (size_t)NN * 8 * 4, stream);
    hipMemsetAsync(attnout, 0, (size_t)NN * 256 * 4, stream);
    attn_score_kernel<<<(EE * 8 + 255) / 256, 256, 0, stream>>>(qbuf, kbuf, edge_attr,
        L.ejw, L.ejb, eidx, scores, smax);
    attn_exp_kernel<<<(EE * 8 + 255) / 256, 256, 0, stream>>>(scores, smax, eidx, exb, denom);
    attn_out_kernel<<<EE, 256, 0, stream>>>(vbuf, exb, denom, eidx, attnout);
    cvt(attnout, ntmp16, NN * 256);
    gemm16_kernel<0><<<gblk(NN, 256), 256, 0, stream>>>(ntmp16, Z.o, L.ob, aggr, nullptr, NN, 256, 256);
    ln_kernel<<<NN, 256, 0, stream>>>(aggr, nullptr, L.alng, L.alnb, h, h16);
  }

  // --- output MLP ---
  gemm16_kernel<1><<<gblk(NN, 256), 256, 0, stream>>>(h16, out1_z, out1_b, nullptr, ntmp16, NN, 256, 256);
  gemm16_kernel<0><<<gblk(NN, 128), 256, 0, stream>>>(ntmp16, out2_z, out2_b, (float*)d_out, nullptr, NN, 256, 128);
}
